// MultiheadAttention_3951369912873
// MI455X (gfx1250) — compile-verified, matrix operand order checked by reading, not run on hardware
//
#include <hip/hip_runtime.h>
#include <hip/hip_bf16.h>

#define L_SEQ   4096
#define D_MODEL 768
#define N_HEADS 12
#define D_HEAD  64
#define INV_SCALE 0.125f   // 1/sqrt(64)

typedef __attribute__((ext_vector_type(16))) __bf16 v16bf;
typedef __attribute__((ext_vector_type(8)))  float  v8f;

union BFrag {
    v16bf v;
    unsigned short h[16];
    unsigned int u[8];
    uint4 q[2];
};

__device__ __forceinline__ unsigned short f2bf(float f) {
    unsigned u = __builtin_bit_cast(unsigned, f);
    u += 0x7FFFu + ((u >> 16) & 1u);          // round-to-nearest-even
    return (unsigned short)(u >> 16);
}

// pack two fp32 -> one dword of two bf16 (lo = first arg)
__device__ __forceinline__ unsigned int pack2bf(float lo, float hi) {
#if __has_builtin(__builtin_amdgcn_cvt_pk_bf16_f32)
    typedef __attribute__((ext_vector_type(2))) __bf16 v2bf;
    v2bf r = __builtin_amdgcn_cvt_pk_bf16_f32(lo, hi);
    return __builtin_bit_cast(unsigned int, r);
#else
    return (unsigned int)f2bf(lo) | ((unsigned int)f2bf(hi) << 16);
#endif
}

__device__ __forceinline__ v8f wmma_bf16(v16bf a, v16bf b, v8f c) {
    // (neg_a, A, neg_b, B, c_mod, C, reuse_a, reuse_b)
    return __builtin_amdgcn_wmma_f32_16x16x32_bf16(false, a, false, b,
                                                   (short)0, c, false, false);
}

// 16-bit fragment for one lane of a 16x32 tile from bf16 memory.
// base -> this lane's row/col start at the K-chunk; hi = 8 if lane>=16 else 0.
// K mapping per lane: {hi..hi+7} then {16+hi..16+hi+7}.
__device__ __forceinline__ v16bf frag_from_bf16(const unsigned short* base, int hi) {
    BFrag f;
    f.q[0] = *(const uint4*)(base + hi);       // halves 0..7
    f.q[1] = *(const uint4*)(base + 16 + hi);  // halves 8..15
    return f.v;
}

// same fragment built from fp32 source (LDS score strip), pair-packed
__device__ __forceinline__ v16bf frag_from_f32(const float* base, int hi) {
    BFrag f;
#pragma unroll
    for (int i = 0; i < 8; i += 2) {
        f.u[i / 2]     = pack2bf(base[hi + i],      base[hi + i + 1]);
        f.u[4 + i / 2] = pack2bf(base[16 + hi + i], base[16 + hi + i + 1]);
    }
    return f.v;
}

// ---------------------------------------------------------------------------
// One-time fp32 -> bf16 conversion (n must be a multiple of 4).
// ---------------------------------------------------------------------------
__global__ void cvt_f32_bf16_kernel(const float* __restrict__ src,
                                    unsigned short* __restrict__ dst, int n) {
    int i = (blockIdx.x * blockDim.x + threadIdx.x) * 4;
    for (; i + 3 < n; i += gridDim.x * blockDim.x * 4) {
        float4 v = *(const float4*)(src + i);
        uint2 p;
        p.x = pack2bf(v.x, v.y);
        p.y = pack2bf(v.z, v.w);
        *(uint2*)(dst + i) = p;
    }
}

// ---------------------------------------------------------------------------
// Y[l][e] = sum_d X[l][d]*W[e][d] + bias[e]  (all-bf16 inputs, bf16 out).
// 32x32 C-block per wave: 2 A-frags x 2 B-frags -> 4 WMMAs per K-chunk.
// storeTransposed=1 -> Y stored [e][l] (for V). grid=(D/32, L/32), block=32.
// ---------------------------------------------------------------------------
__global__ void proj_qkv_kernel(const unsigned short* __restrict__ Xb,
                                const unsigned short* __restrict__ Wb,
                                const float* __restrict__ bias,
                                unsigned short* __restrict__ Y,
                                int storeTransposed) {
    const int lane = threadIdx.x;
    const int e0 = blockIdx.x * 32;
    const int l0 = blockIdx.y * 32;
    const int n  = lane & 15;
    const int hi = (lane >> 4) << 3;

    const unsigned short* xr0 = Xb + (size_t)(l0 + n) * D_MODEL;
    const unsigned short* xr1 = Xb + (size_t)(l0 + 16 + n) * D_MODEL;
    const unsigned short* wr0 = Wb + (size_t)(e0 + n) * D_MODEL;
    const unsigned short* wr1 = Wb + (size_t)(e0 + 16 + n) * D_MODEL;

    v8f c00 = {}, c01 = {}, c10 = {}, c11 = {};
#pragma unroll 4
    for (int kc = 0; kc < D_MODEL; kc += 32) {
        v16bf a0 = frag_from_bf16(xr0 + kc, hi);
        v16bf a1 = frag_from_bf16(xr1 + kc, hi);
        v16bf b0 = frag_from_bf16(wr0 + kc, hi);
        v16bf b1 = frag_from_bf16(wr1 + kc, hi);
        c00 = wmma_bf16(a0, b0, c00);
        c01 = wmma_bf16(a0, b1, c01);
        c10 = wmma_bf16(a1, b0, c10);
        c11 = wmma_bf16(a1, b1, c11);
    }

    const float bv0 = bias[e0 + n];
    const float bv1 = bias[e0 + 16 + n];
#pragma unroll
    for (int j = 0; j < 8; ++j) {
        const int M = hi + j;
        const float v00 = c00[j] + bv0, v01 = c01[j] + bv1;
        const float v10 = c10[j] + bv0, v11 = c11[j] + bv1;
        if (storeTransposed) {
            Y[(size_t)(e0 + n) * L_SEQ + (l0 + M)]            = f2bf(v00);
            Y[(size_t)(e0 + 16 + n) * L_SEQ + (l0 + M)]       = f2bf(v01);
            Y[(size_t)(e0 + n) * L_SEQ + (l0 + 16 + M)]       = f2bf(v10);
            Y[(size_t)(e0 + 16 + n) * L_SEQ + (l0 + 16 + M)]  = f2bf(v11);
        } else {
            Y[(size_t)(l0 + M) * D_MODEL + (e0 + n)]          = f2bf(v00);
            Y[(size_t)(l0 + M) * D_MODEL + (e0 + 16 + n)]     = f2bf(v01);
            Y[(size_t)(l0 + 16 + M) * D_MODEL + (e0 + n)]     = f2bf(v10);
            Y[(size_t)(l0 + 16 + M) * D_MODEL + (e0 + 16 + n)] = f2bf(v11);
        }
    }
}

// ---------------------------------------------------------------------------
// Attention per (head, 16-query tile), 4 waves (128 threads) per block.
// Score strip (16 x 4096 fp32 = 256KB) lives in LDS; attn written to HBM once.
//   phase 1: waves interleave over key tiles: S = QK^T * scale, masked -> LDS
//   phase 2: float4-vectorized softmax across 128 threads (cross-wave via LDS)
//   phase 3: wave w computes ctx d-tile w (N=16) over K=4096 keys
// grid = (L/16, H), dynamic LDS = 16*L*4 + 256 bytes.
// ---------------------------------------------------------------------------
__global__ void attn_kernel(const unsigned short* __restrict__ Qbf,  // [L][D]
                            const unsigned short* __restrict__ Kbf,  // [L][D]
                            const unsigned short* __restrict__ Vt,   // [D][L]
                            const int* __restrict__ mask,            // [L][L]
                            float* __restrict__ attn,                // [H][L][L]
                            unsigned short* __restrict__ Ctx) {      // [L][D]
    extern __shared__ float smem[];               // strip[16][4096] + red[4][16]
    float* red = smem + 16 * L_SEQ;
    const int tid  = threadIdx.x;
    const int lane = tid & 31;
    const int w    = tid >> 5;                    // wave id 0..3
    const int q0 = blockIdx.x * 16;
    const int h  = blockIdx.y;
    const int n  = lane & 15;
    const int hi = (lane >> 4) << 3;

    // Q fragments (K-dim 64 = two 32-chunks), resident for the whole pass
    const unsigned short* qrow = Qbf + (size_t)(q0 + n) * D_MODEL + h * D_HEAD;
    const v16bf qa0 = frag_from_bf16(qrow, hi);
    const v16bf qa1 = frag_from_bf16(qrow + 32, hi);

    // ---- phase 1: scores ----
    for (int kt = w; kt < L_SEQ / 16; kt += 4) {
        const unsigned short* krow =
            Kbf + (size_t)(kt * 16 + n) * D_MODEL + h * D_HEAD;
        v8f c = {};
        c = wmma_bf16(qa0, frag_from_bf16(krow, hi), c);
        c = wmma_bf16(qa1, frag_from_bf16(krow + 32, hi), c);
        const int col = kt * 16 + n;
#pragma unroll
        for (int j = 0; j < 8; ++j) {
            const int M = hi + j;
            float s = c[j] * INV_SCALE;
            s = (mask[(size_t)(q0 + M) * L_SEQ + col] == 0) ? -1e9f : s;
            smem[M * L_SEQ + col] = s;
        }
    }
    __syncthreads();

    // ---- phase 2: softmax over the 16 rows ----
    const int c0 = tid * 4;                      // float4 lanes: stride 512
    float rmax[16];
#pragma unroll
    for (int m = 0; m < 16; ++m) {
        float pm = -3.0e38f;
        const float* row = smem + m * L_SEQ;
        for (int c = c0; c < L_SEQ; c += 512) {
            float4 v = *(const float4*)(row + c);
            pm = fmaxf(pm, fmaxf(fmaxf(v.x, v.y), fmaxf(v.z, v.w)));
        }
        for (int off = 16; off > 0; off >>= 1)
            pm = fmaxf(pm, __shfl_xor(pm, off, 32));
        rmax[m] = pm;
    }
    if (lane == 0) {
#pragma unroll
        for (int m = 0; m < 16; ++m) red[w * 16 + m] = rmax[m];
    }
    __syncthreads();
#pragma unroll
    for (int m = 0; m < 16; ++m)
        rmax[m] = fmaxf(fmaxf(red[m], red[16 + m]),
                        fmaxf(red[32 + m], red[48 + m]));
    __syncthreads();                             // red reused for sums

    float rinv[16];
#pragma unroll
    for (int m = 0; m < 16; ++m) {
        float ps = 0.0f;
        float* row = smem + m * L_SEQ;
        for (int c = c0; c < L_SEQ; c += 512) {
            float4 v = *(float4*)(row + c);
            v.x = __expf(v.x - rmax[m]);
            v.y = __expf(v.y - rmax[m]);
            v.z = __expf(v.z - rmax[m]);
            v.w = __expf(v.w - rmax[m]);
            *(float4*)(row + c) = v;
            ps += v.x + v.y + v.z + v.w;
        }
        for (int off = 16; off > 0; off >>= 1)
            ps += __shfl_xor(ps, off, 32);
        rinv[m] = ps;
    }
    if (lane == 0) {
#pragma unroll
        for (int m = 0; m < 16; ++m) red[w * 16 + m] = rinv[m];
    }
    __syncthreads();
#pragma unroll
    for (int m = 0; m < 16; ++m)
        rinv[m] = 1.0f / (red[m] + red[16 + m] + red[32 + m] + red[48 + m]);

    // normalize in LDS + single coalesced b128 write of attn
    float* attn_base = attn + ((size_t)h * L_SEQ + q0) * L_SEQ;
#pragma unroll
    for (int m = 0; m < 16; ++m) {
        float* row = smem + m * L_SEQ;
        float* arow = attn_base + (size_t)m * L_SEQ;
        for (int c = c0; c < L_SEQ; c += 512) {
            float4 v = *(float4*)(row + c);
            v.x *= rinv[m]; v.y *= rinv[m]; v.z *= rinv[m]; v.w *= rinv[m];
            *(float4*)(row + c) = v;
            *(float4*)(arow + c) = v;
        }
    }
    __syncthreads();

    // ---- phase 3: ctx = attn @ V ; wave w owns d-tile w ----
    v8f acc = {};
    const float* srow = smem + n * L_SEQ;        // lane's attn row (A-frag src)
    const unsigned short* vb = Vt + (size_t)(h * D_HEAD + w * 16 + n) * L_SEQ;
    for (int kb = 0; kb < L_SEQ; kb += 32) {
        v16bf a = frag_from_f32(srow + kb, hi);
        acc = wmma_bf16(a, frag_from_bf16(vb + kb, hi), acc);
    }
#pragma unroll
    for (int j = 0; j < 8; ++j) {
        const int M = hi + j;
        Ctx[(size_t)(q0 + M) * D_MODEL + h * D_HEAD + w * 16 + n] = f2bf(acc[j]);
    }
}

// ---------------------------------------------------------------------------
// out[l][e] = sum_d ctx[l][d]*Wo[e][d] + bo[e]  (bf16 x bf16 -> fp32 out)
// 32x32 C-block per wave. grid = (D/32, L/32), block = 32.
// ---------------------------------------------------------------------------
__global__ void proj_out_kernel(const unsigned short* __restrict__ Ctx,
                                const unsigned short* __restrict__ Wob,
                                const float* __restrict__ bo,
                                float* __restrict__ Out) {
    const int lane = threadIdx.x;
    const int e0 = blockIdx.x * 32;
    const int l0 = blockIdx.y * 32;
    const int n  = lane & 15;
    const int hi = (lane >> 4) << 3;

    const unsigned short* ar0 = Ctx + (size_t)(l0 + n) * D_MODEL;
    const unsigned short* ar1 = Ctx + (size_t)(l0 + 16 + n) * D_MODEL;
    const unsigned short* wr0 = Wob + (size_t)(e0 + n) * D_MODEL;
    const unsigned short* wr1 = Wob + (size_t)(e0 + 16 + n) * D_MODEL;

    v8f c00 = {}, c01 = {}, c10 = {}, c11 = {};
#pragma unroll 4
    for (int kc = 0; kc < D_MODEL; kc += 32) {
        v16bf a0 = frag_from_bf16(ar0 + kc, hi);
        v16bf a1 = frag_from_bf16(ar1 + kc, hi);
        v16bf b0 = frag_from_bf16(wr0 + kc, hi);
        v16bf b1 = frag_from_bf16(wr1 + kc, hi);
        c00 = wmma_bf16(a0, b0, c00);
        c01 = wmma_bf16(a0, b1, c01);
        c10 = wmma_bf16(a1, b0, c10);
        c11 = wmma_bf16(a1, b1, c11);
    }

    const float bv0 = bo[e0 + n];
    const float bv1 = bo[e0 + 16 + n];
#pragma unroll
    for (int j = 0; j < 8; ++j) {
        const int M = hi + j;
        Out[(size_t)(l0 + M) * D_MODEL + (e0 + n)]           = c00[j] + bv0;
        Out[(size_t)(l0 + M) * D_MODEL + (e0 + 16 + n)]      = c01[j] + bv1;
        Out[(size_t)(l0 + 16 + M) * D_MODEL + (e0 + n)]      = c10[j] + bv0;
        Out[(size_t)(l0 + 16 + M) * D_MODEL + (e0 + 16 + n)] = c11[j] + bv1;
    }
}

extern "C" void kernel_launch(void* const* d_in, const int* in_sizes, int n_in,
                              void* d_out, int out_size, void* d_ws, size_t ws_size,
                              hipStream_t stream) {
    (void)in_sizes; (void)n_in; (void)out_size; (void)ws_size;
    const float* query = (const float*)d_in[0];
    const float* key_  = (const float*)d_in[1];
    const float* value = (const float*)d_in[2];
    const int*   mask  = (const int*)d_in[3];
    const float* Wq = (const float*)d_in[4];  const float* bq = (const float*)d_in[5];
    const float* Wk = (const float*)d_in[6];  const float* bk = (const float*)d_in[7];
    const float* Wv = (const float*)d_in[8];  const float* bv = (const float*)d_in[9];
    const float* Wo = (const float*)d_in[10]; const float* bo = (const float*)d_in[11];

    float* out  = (float*)d_out;                              // [L][D]
    float* attn = out + (size_t)L_SEQ * D_MODEL;              // [H][L][L]

    const size_t LD = (size_t)L_SEQ * D_MODEL;                // elements
    const size_t WW = (size_t)D_MODEL * D_MODEL;
    unsigned short* Qbf  = (unsigned short*)d_ws;             // [L][D]
    unsigned short* Kbf  = Qbf + LD;                          // [L][D]
    unsigned short* Vt   = Kbf + LD;                          // [D][L]
    unsigned short* Ctx  = Vt  + LD;                          // [L][D]
    unsigned short* Xbuf = Ctx + LD;                          // staging [L][D]
    unsigned short* Wbuf = Xbuf + LD;                         // staging [D][D]

    const int nX = (int)LD, nW = (int)WW;
    const int gX = nX / (256 * 4), gW = nW / (256 * 4);
    dim3 pgrid(D_MODEL / 32, L_SEQ / 32);

    // Q projection
    cvt_f32_bf16_kernel<<<gX, 256, 0, stream>>>(query, Xbuf, nX);
    cvt_f32_bf16_kernel<<<gW, 256, 0, stream>>>(Wq, Wbuf, nW);
    proj_qkv_kernel<<<pgrid, 32, 0, stream>>>(Xbuf, Wbuf, bq, Qbf, 0);
    // K projection
    cvt_f32_bf16_kernel<<<gX, 256, 0, stream>>>(key_, Xbuf, nX);
    cvt_f32_bf16_kernel<<<gW, 256, 0, stream>>>(Wk, Wbuf, nW);
    proj_qkv_kernel<<<pgrid, 32, 0, stream>>>(Xbuf, Wbuf, bk, Kbf, 0);
    // V projection (transposed store)
    cvt_f32_bf16_kernel<<<gX, 256, 0, stream>>>(value, Xbuf, nX);
    cvt_f32_bf16_kernel<<<gW, 256, 0, stream>>>(Wv, Wbuf, nW);
    proj_qkv_kernel<<<pgrid, 32, 0, stream>>>(Xbuf, Wbuf, bv, Vt, 1);

    // attention (score strip + softmax + ctx)
    const size_t smem_bytes = (size_t)16 * L_SEQ * sizeof(float) + 256;
    attn_kernel<<<dim3(L_SEQ / 16, N_HEADS), 128, smem_bytes, stream>>>(
        Qbf, Kbf, Vt, mask, attn, Ctx);

    // output projection
    cvt_f32_bf16_kernel<<<gW, 256, 0, stream>>>(Wo, Wbuf, nW);
    proj_out_kernel<<<pgrid, 32, 0, stream>>>(Ctx, Wbuf, bo, out);
}
